// MultiheadedAttention_1675037246169
// MI455X (gfx1250) — compile-verified, matrix operand order checked by reading, not run on hardware
//
#include <hip/hip_runtime.h>
#include <hip/hip_bf16.h>

typedef __attribute__((ext_vector_type(16))) _Float16 v16h;
typedef __attribute__((ext_vector_type(8)))  _Float16 h8;
typedef __attribute__((ext_vector_type(8)))  float    v8f;

#define SQ   4096        // sequence length S
#define DM   512         // model dim D
#define NROW 16384       // B*S
#define NB   4           // 16x64 tile per wave -> 4 wmma accumulators

// ---- WMMA helper: D = A(16x32 f16) x B(32x16 f16) + C(f32) --------------
__device__ __forceinline__ v8f wmma_f16(v16h a, v16h b, v8f c) {
  return __builtin_amdgcn_wmma_f32_16x16x32_f16(
      /*neg_a=*/false, a, /*neg_b=*/false, b,
      /*c_mod=*/(short)0, c, /*reuse_a=*/false, /*reuse_b=*/false);
}

// Fragment loader, f16 source, row-major with K along the row.
// Lane layout (ISA 7.12.2, 16-bit A 16x32): lane holds row (lane&15);
// v16h elems 0..7 = K[kh*8 .. kh*8+7], elems 8..15 = K[16+kh*8 ..], kh=lane>>4.
__device__ __forceinline__ v16h ld_frag_f16(const _Float16* p, int kh) {
  h8 lo = *(const h8*)(p + kh * 8);
  h8 hi = *(const h8*)(p + 16 + kh * 8);
  v16h r;
#pragma unroll
  for (int i = 0; i < 8; ++i) { r[i] = lo[i]; r[i + 8] = hi[i]; }
  return r;
}

// Same fragment, fp32 source converted to f16 on the fly.
__device__ __forceinline__ v16h ld_frag_f32(const float* p, int kh) {
  float4 a0 = *(const float4*)(p + kh * 8);
  float4 a1 = *(const float4*)(p + kh * 8 + 4);
  float4 b0 = *(const float4*)(p + 16 + kh * 8);
  float4 b1 = *(const float4*)(p + 16 + kh * 8 + 4);
  v16h r;
  r[0]  = (_Float16)a0.x; r[1]  = (_Float16)a0.y; r[2]  = (_Float16)a0.z; r[3]  = (_Float16)a0.w;
  r[4]  = (_Float16)a1.x; r[5]  = (_Float16)a1.y; r[6]  = (_Float16)a1.z; r[7]  = (_Float16)a1.w;
  r[8]  = (_Float16)b0.x; r[9]  = (_Float16)b0.y; r[10] = (_Float16)b0.z; r[11] = (_Float16)b0.w;
  r[12] = (_Float16)b1.x; r[13] = (_Float16)b1.y; r[14] = (_Float16)b1.z; r[15] = (_Float16)b1.w;
  return r;
}

// Async copy of 16 bytes global -> LDS (ASYNCcnt-tracked, per-lane addresses).
__device__ __forceinline__ void async_g2l_b128(unsigned lds_off, const void* gaddr) {
  asm volatile("global_load_async_to_lds_b128 %0, %1, off"
               :: "v"(lds_off), "v"((unsigned long long)gaddr) : "memory");
}
__device__ __forceinline__ void wait_async0() {
  asm volatile("s_wait_asynccnt 0x0" ::: "memory");
}

// ---- Kernel 1: q/k/v = x @ W^T + b ; q,k -> f16 row-major, v -> f16 transposed
__global__ void k_qkv(const float* __restrict__ x,
                      const float* __restrict__ Wq, const float* __restrict__ Wk,
                      const float* __restrict__ Wv,
                      const float* __restrict__ bq, const float* __restrict__ bk,
                      const float* __restrict__ bv,
                      _Float16* __restrict__ q16, _Float16* __restrict__ k16,
                      _Float16* __restrict__ vT) {
  const int tid  = threadIdx.x;
  const int lane = tid & 31, ln = lane & 15, kh = lane >> 4;
  const int gw   = blockIdx.x * 8 + (tid >> 5);      // 24576 waves total
  const int mat  = gw / 8192;                        // 0=q 1=k 2=v
  const int rem  = gw % 8192;
  const int m0   = (rem >> 3) * 16;                  // global row block (B*S)
  const int n0   = (rem & 7) * 64;                   // output column block
  const float* W  = (mat == 0) ? Wq : (mat == 1) ? Wk : Wv;
  const float* bi = (mat == 0) ? bq : (mat == 1) ? bk : bv;

  v8f acc[NB] = {};
  const float* arow = x + (size_t)(m0 + ln) * DM;
#pragma unroll 1
  for (int kk = 0; kk < DM; kk += 32) {
    v16h a = ld_frag_f32(arow + kk, kh);
#pragma unroll
    for (int t = 0; t < NB; ++t) {
      v16h b = ld_frag_f32(W + (size_t)(n0 + t * 16 + ln) * DM + kk, kh);
      acc[t] = wmma_f16(a, b, acc[t]);
    }
  }
  if (mat < 2) {
    _Float16* dst = (mat == 0) ? q16 : k16;
#pragma unroll
    for (int t = 0; t < NB; ++t) {
      const int col = n0 + t * 16 + ln;
      const float bb = bi[col];
#pragma unroll
      for (int j = 0; j < 8; ++j)
        dst[(size_t)(m0 + j + 8 * kh) * DM + col] = (_Float16)(acc[t][j] + bb);
    }
  } else {
    // transposed store: vT[b][d][s], contiguous 8 halfs per lane
    const int bidx = m0 >> 12;
    const int s0   = m0 & (SQ - 1);
#pragma unroll
    for (int t = 0; t < NB; ++t) {
      const int d = n0 + t * 16 + ln;
      const float bb = bi[d];
      h8 pk;
#pragma unroll
      for (int j = 0; j < 8; ++j) pk[j] = (_Float16)(acc[t][j] + bb);
      *(h8*)(vT + ((size_t)bidx * DM + d) * SQ + s0 + kh * 8) = pk;
    }
  }
}

// ---- Kernel 2: raw scores = (q @ k^T) / 8 -> attn region (fp32) -----------
// All 8 waves of a block share the same (b, m0) A panel: stage 16x512 f16
// once via async-to-LDS, then the whole K loop reads A from LDS.
__global__ void k_scores(const _Float16* __restrict__ q16,
                         const _Float16* __restrict__ k16,
                         float* __restrict__ attn) {
  const int tid  = threadIdx.x;
  const int lane = tid & 31, ln = lane & 15, kh = lane >> 4;
  const int gw   = blockIdx.x * 8 + (tid >> 5);      // 65536 waves
  const int b    = gw >> 14;                         // 16384 waves / batch
  const int rem  = gw & 16383;
  const int m0   = (rem >> 6) * 16;                  // query rows (uniform per block)
  const int n0   = (rem & 63) * 64;                  // key cols per wave

  __shared__ _Float16 As[16 * DM];                   // 16 KB A panel
  {
    const int r = tid >> 4;                          // row 0..15
    const int c = (tid & 15) * 32;                   // 32 halfs = 64B per thread
    const _Float16* g = q16 + (size_t)(b * SQ + m0 + r) * DM + c;
    unsigned l = (unsigned)(size_t)&As[r * DM + c];
#pragma unroll
    for (int i = 0; i < 4; ++i)
      async_g2l_b128(l + i * 16, (const char*)g + i * 16);
    wait_async0();
  }
  __syncthreads();

  v8f acc[NB] = {};
#pragma unroll 1
  for (int kk = 0; kk < DM; kk += 32) {
    v16h a;
    {
      h8 lo = *(const h8*)&As[ln * DM + kk + kh * 8];
      h8 hi = *(const h8*)&As[ln * DM + kk + 16 + kh * 8];
#pragma unroll
      for (int i = 0; i < 8; ++i) { a[i] = lo[i]; a[i + 8] = hi[i]; }
    }
#pragma unroll
    for (int t = 0; t < NB; ++t) {
      v16h bm = ld_frag_f16(k16 + (size_t)(b * SQ + n0 + t * 16 + ln) * DM + kk, kh);
      acc[t] = wmma_f16(a, bm, acc[t]);
    }
  }
#pragma unroll
  for (int t = 0; t < NB; ++t) {
    const int col = n0 + t * 16 + ln;
#pragma unroll
    for (int j = 0; j < 8; ++j)
      attn[(size_t)(b * SQ + m0 + j + 8 * kh) * SQ + col] = acc[t][j] * 0.125f;
  }
}

// ---- Kernel 3: in-place row softmax over attn (one block per row) ---------
__global__ void k_softmax(float* __restrict__ attn) {
  const int tid = threadIdx.x;
  float* row = attn + (size_t)blockIdx.x * SQ;
  float4 v[4];
#pragma unroll
  for (int i = 0; i < 4; ++i) v[i] = *((const float4*)row + tid + i * 256);

  float m = -3.0e38f;
#pragma unroll
  for (int i = 0; i < 4; ++i)
    m = fmaxf(m, fmaxf(fmaxf(v[i].x, v[i].y), fmaxf(v[i].z, v[i].w)));
#pragma unroll
  for (int off = 1; off < 32; off <<= 1) m = fmaxf(m, __shfl_xor(m, off, 32));

  __shared__ float smax[8], ssum[8];
  if ((tid & 31) == 0) smax[tid >> 5] = m;
  __syncthreads();
  m = smax[0];
#pragma unroll
  for (int w = 1; w < 8; ++w) m = fmaxf(m, smax[w]);

  float s = 0.f;
#pragma unroll
  for (int i = 0; i < 4; ++i) {
    v[i].x = __expf(v[i].x - m); v[i].y = __expf(v[i].y - m);
    v[i].z = __expf(v[i].z - m); v[i].w = __expf(v[i].w - m);
    s += v[i].x + v[i].y + v[i].z + v[i].w;
  }
#pragma unroll
  for (int off = 1; off < 32; off <<= 1) s += __shfl_xor(s, off, 32);
  if ((tid & 31) == 0) ssum[tid >> 5] = s;
  __syncthreads();
  s = 0.f;
#pragma unroll
  for (int w = 0; w < 8; ++w) s += ssum[w];
  const float inv = 1.0f / s;
#pragma unroll
  for (int i = 0; i < 4; ++i) {
    v[i].x *= inv; v[i].y *= inv; v[i].z *= inv; v[i].w *= inv;
    *((float4*)row + tid + i * 256) = v[i];
  }
}

// ---- Kernel 4: ctx = attn @ v  (B operand = rows of vT) -------------------
__global__ void k_context(const float* __restrict__ attn,
                          const _Float16* __restrict__ vT,
                          _Float16* __restrict__ ctx) {
  const int tid  = threadIdx.x;
  const int lane = tid & 31, ln = lane & 15, kh = lane >> 4;
  const int gw   = blockIdx.x * 8 + (tid >> 5);      // 8192 waves
  const int m0   = (gw >> 3) * 16;                   // global row (B*S)
  const int n0   = (gw & 7) * 64;                    // D columns
  const int b    = m0 >> 12;

  v8f acc[NB] = {};
  const float* arow = attn + (size_t)(m0 + ln) * SQ;
  const _Float16* vb = vT + (size_t)b * DM * SQ;
#pragma unroll 1
  for (int kk = 0; kk < SQ; kk += 32) {
    // keep the HBM pipe full on the big streaming read (268 MB of attn)
    __builtin_prefetch(arow + kk + 128, 0, 1);
    v16h a = ld_frag_f32(arow + kk, kh);
#pragma unroll
    for (int t = 0; t < NB; ++t) {
      v16h bm = ld_frag_f16(vb + (size_t)(n0 + t * 16 + ln) * SQ + kk, kh);
      acc[t] = wmma_f16(a, bm, acc[t]);
    }
  }
#pragma unroll
  for (int t = 0; t < NB; ++t) {
    const int col = n0 + t * 16 + ln;
#pragma unroll
    for (int j = 0; j < 8; ++j)
      ctx[(size_t)(m0 + j + 8 * kh) * DM + col] = (_Float16)acc[t][j];
  }
}

// ---- Kernel 5: out = LayerNorm(ctx @ Wo^T + bo + x) -----------------------
// Same shared-A structure as k_scores: stage ctx row panel via async-to-LDS.
__global__ void k_projln(const _Float16* __restrict__ ctx,
                         const float* __restrict__ Wo, const float* __restrict__ bo,
                         const float* __restrict__ x,
                         const float* __restrict__ gamma, const float* __restrict__ beta,
                         float* __restrict__ out) {
  const int tid  = threadIdx.x;
  const int lane = tid & 31, ln = lane & 15, kh = lane >> 4;
  const int wv   = tid >> 5;                         // 8 waves cover 8*64 = 512 cols
  const int m0   = blockIdx.x * 16;                  // 16 rows per block
  const int n0   = wv * 64;

  __shared__ _Float16 As[16 * DM];                   // 16 KB A panel
  {
    const int r = tid >> 4;
    const int c = (tid & 15) * 32;
    const _Float16* g = ctx + (size_t)(m0 + r) * DM + c;
    unsigned l = (unsigned)(size_t)&As[r * DM + c];
#pragma unroll
    for (int i = 0; i < 4; ++i)
      async_g2l_b128(l + i * 16, (const char*)g + i * 16);
    wait_async0();
  }
  __syncthreads();

  v8f acc[NB] = {};
#pragma unroll 1
  for (int kk = 0; kk < DM; kk += 32) {
    v16h a;
    {
      h8 lo = *(const h8*)&As[ln * DM + kk + kh * 8];
      h8 hi = *(const h8*)&As[ln * DM + kk + 16 + kh * 8];
#pragma unroll
      for (int i = 0; i < 8; ++i) { a[i] = lo[i]; a[i + 8] = hi[i]; }
    }
#pragma unroll
    for (int t = 0; t < NB; ++t) {
      v16h bm = ld_frag_f32(Wo + (size_t)(n0 + t * 16 + ln) * DM + kk, kh);
      acc[t] = wmma_f16(a, bm, acc[t]);
    }
  }

  float val[NB][8];
  float s[8], s2[8];
#pragma unroll
  for (int j = 0; j < 8; ++j) { s[j] = 0.f; s2[j] = 0.f; }
#pragma unroll
  for (int t = 0; t < NB; ++t) {
    const int col = n0 + t * 16 + ln;
    const float bb = bo[col];
#pragma unroll
    for (int j = 0; j < 8; ++j) {
      float f = acc[t][j] + bb + x[(size_t)(m0 + j + 8 * kh) * DM + col];
      val[t][j] = f;
      s[j] += f; s2[j] += f * f;
    }
  }
  // per-row reduce across the 16 lanes of this half (masks 1,2,4,8 stay in-half)
#pragma unroll
  for (int off = 1; off < 16; off <<= 1) {
#pragma unroll
    for (int j = 0; j < 8; ++j) {
      s[j]  += __shfl_xor(s[j],  off, 32);
      s2[j] += __shfl_xor(s2[j], off, 32);
    }
  }
  __shared__ float ps[8][16], pq[8][16], smu[16], sis[16];
  if (ln == 0) {
#pragma unroll
    for (int j = 0; j < 8; ++j) {
      ps[wv][j + 8 * kh] = s[j];
      pq[wv][j + 8 * kh] = s2[j];
    }
  }
  __syncthreads();
  if (tid < 16) {
    float a = 0.f, bsum = 0.f;
#pragma unroll
    for (int w = 0; w < 8; ++w) { a += ps[w][tid]; bsum += pq[w][tid]; }
    const float m  = a * (1.0f / 512.0f);
    const float vr = bsum * (1.0f / 512.0f) - m * m;
    smu[tid] = m;
    sis[tid] = rsqrtf(vr + 1e-5f);
  }
  __syncthreads();
#pragma unroll
  for (int t = 0; t < NB; ++t) {
    const int col = n0 + t * 16 + ln;
    const float g = gamma[col], be = beta[col];
#pragma unroll
    for (int j = 0; j < 8; ++j) {
      const int M = j + 8 * kh;
      out[(size_t)(m0 + M) * DM + col] = (val[t][j] - smu[M]) * sis[M] * g + be;
    }
  }
}

// ---- host-side launch ------------------------------------------------------
extern "C" void kernel_launch(void* const* d_in, const int* in_sizes, int n_in,
                              void* d_out, int out_size, void* d_ws, size_t ws_size,
                              hipStream_t stream) {
  const float* x     = (const float*)d_in[0];
  const float* Wq    = (const float*)d_in[1];
  const float* bq    = (const float*)d_in[2];
  const float* Wk    = (const float*)d_in[3];
  const float* bk    = (const float*)d_in[4];
  const float* Wv    = (const float*)d_in[5];
  const float* bv    = (const float*)d_in[6];
  const float* Wo    = (const float*)d_in[7];
  const float* bo    = (const float*)d_in[8];
  const float* gamma = (const float*)d_in[9];
  const float* beta  = (const float*)d_in[10];

  float* out  = (float*)d_out;                 // [B,S,D]
  float* attn = out + (size_t)NROW * DM;       // [B,S,S]

  _Float16* q16 = (_Float16*)d_ws;                 // 16 MB each
  _Float16* k16 = q16 + (size_t)NROW * DM;
  _Float16* vT  = k16 + (size_t)NROW * DM;         // [B, D, S]
  _Float16* ctx = vT  + (size_t)NROW * DM;

  k_qkv    <<<3072, 256, 0, stream>>>(x, Wq, Wk, Wv, bq, bk, bv, q16, k16, vT);
  k_scores <<<8192, 256, 0, stream>>>(q16, k16, attn);
  k_softmax<<<16384, 256, 0, stream>>>(attn);
  k_context<<<1024, 256, 0, stream>>>(attn, vT, ctx);
  k_projln <<<1024, 256, 0, stream>>>(ctx, Wo, bo, x, gamma, beta, out);
}
